// KendallsTauLoss_24438363914244
// MI455X (gfx1250) — 8-bit matrix operand order repaired and checked by reading, compiled, not run on hardware
//
#include <hip/hip_runtime.h>
#include <hip/hip_bf16.h>

typedef __attribute__((ext_vector_type(8))) int v8i;

#define JSLICE 2048

__device__ __forceinline__ unsigned pack_signs4(float a, float4 x) {
    int s0 = (a > x.x) - (a < x.x);
    int s1 = (a > x.y) - (a < x.y);
    int s2 = (a > x.z) - (a < x.z);
    int s3 = (a > x.w) - (a < x.w);
    unsigned t01 = __builtin_amdgcn_perm((unsigned)s1, (unsigned)s0, 0x00000400u);
    unsigned t23 = __builtin_amdgcn_perm((unsigned)s3, (unsigned)s2, 0x00000400u);
    return __builtin_amdgcn_perm(t23, t01, 0x05040100u);
}

__global__ __launch_bounds__(256)
void kendall_wmma_kernel(const float* __restrict__ p, const float* __restrict__ t,
                         int* __restrict__ accum, int n) {
    __shared__ float spj[JSLICE];
    __shared__ float stj[JSLICE];

    const int tid      = threadIdx.x;
    const int nslices  = n / JSLICE;
    const int blksPerS = gridDim.x / nslices;
    const int slice    = blockIdx.x / blksPerS;
    const int blkInS   = blockIdx.x % blksPerS;
    const int jbase    = slice * JSLICE;

    for (int idx = tid * 4; idx < JSLICE; idx += blockDim.x * 4) {
        unsigned lds_p = (unsigned)(unsigned long long)&spj[idx];
        unsigned lds_t = (unsigned)(unsigned long long)&stj[idx];
        const float* gp = &p[jbase + idx];
        const float* gt = &t[jbase + idx];
        asm volatile("global_load_async_to_lds_b128 %0, %1, off"
                     :: "v"(lds_p), "v"(gp) : "memory");
        asm volatile("global_load_async_to_lds_b128 %0, %1, off"
                     :: "v"(lds_t), "v"(gt) : "memory");
    }
    asm volatile("s_wait_asynccnt 0x0" ::: "memory");
    __syncthreads();

    const int wave = tid >> 5;
    const int lid  = tid & 31;
    const int h    = lid >> 4;
    const int m    = lid & 15;
    const int i0   = (blkInS * 8 + wave) * 16;

    const float pi = p[i0 + m];
    const float ti = t[i0 + m];

    v8i acc = {};
    for (int j0 = 0; j0 < JSLICE; j0 += 64) {
        v8i a, b;

        #pragma unroll
        for (int g = 0; g < 4; ++g) {
            const int kb = j0 + g * 16 + h * 8;
            float4 x0 = *(const float4*)&spj[kb];
            float4 x1 = *(const float4*)&spj[kb + 4];
            a[2 * g]     = (int)pack_signs4(pi, x0);
            a[2 * g + 1] = (int)pack_signs4(pi, x1);
        }

        #pragma unroll
        for (int g2 = 0; g2 < 2; ++g2) {
            const int kb = j0 + g2 * 32 + h * 8;
            #pragma unroll
            for (int q = 0; q < 4; ++q) {
                float4 x = *(const float4*)&stj[kb + 4 * (q & 1) + 16 * (q >> 1)];
                b[4 * g2 + q] = (int)pack_signs4(ti, x);
            }
        }

        acc = __builtin_amdgcn_wmma_i32_16x16x64_iu8(true, a, true, b, acc,
                                                     false, false);
    }

    int d = 0;
    #pragma unroll
    for (int r = 0; r < 8; ++r) {
        d += ((lid == r) || (lid == 24 + r)) ? acc[r] : 0;
    }

    d += __builtin_amdgcn_ds_swizzle(d, 0x401F);
    d += __builtin_amdgcn_ds_swizzle(d, 0x201F);
    d += __builtin_amdgcn_ds_swizzle(d, 0x101F);
    d += __builtin_amdgcn_ds_swizzle(d, 0x081F);
    d += __builtin_amdgcn_ds_swizzle(d, 0x041F);
    if (lid == 0) atomicAdd(accum, d);
}

__global__ void kendall_finalize_kernel(const int* __restrict__ accum,
                                        float* __restrict__ out,
                                        long long denom) {
    out[0] = 1.0f - (float)((double)accum[0] / (double)denom);
}

extern "C" void kernel_launch(void* const* d_in, const int* in_sizes, int n_in,
                              void* d_out, int out_size, void* d_ws, size_t ws_size,
                              hipStream_t stream) {
    const float* p = (const float*)d_in[0];
    const float* t = (const float*)d_in[1];
    const int n = in_sizes[0];

    int* accum = (int*)d_ws;
    (void)hipMemsetAsync(accum, 0, sizeof(int), stream);

    const int nslices = n / JSLICE;
    const int blksPerSlice = n / 128;
    dim3 grid(nslices * blksPerSlice);
    dim3 block(256);

    kendall_wmma_kernel<<<grid, block, 0, stream>>>(p, t, accum, n);
    kendall_finalize_kernel<<<1, 1, 0, stream>>>(accum, (float*)d_out,
                                                 (long long)n * (long long)(n - 1));
}
